// AttentionLayers_my_88828513616429
// MI455X (gfx1250) — matrix operand order repaired and checked by reading, not compiled, not run on hardware
//
#include <hip/hip_runtime.h>

// ---------------------------------------------------------------------------
// CDNA5 (gfx1250) windowed-attention transformer.
// bf16 WMMA for all GEMMs; LDS-staged, async-copy double-buffered GEMM.
// ---------------------------------------------------------------------------

typedef __attribute__((ext_vector_type(16))) __bf16 bf16x16;
typedef __attribute__((ext_vector_type(8)))  float  f32x8;
typedef __attribute__((ext_vector_type(4)))  int    v4i;

struct alignas(16) U4 { unsigned x, y, z, w; };
struct alignas(16) F4 { float x, y, z, w; };
union FragU { bf16x16 v; U4 q[2]; };

#define M_TOK   43904      // B*N = 128*343
#define BWIN    128
#define NTOK    343
#define NPAD    352        // 22 * 16
#define DMODEL  512
#define NHEAD   8
#define DHEAD   64
#define DFF     2048
#define NLAYER  6
#define TREL    2197       // 13*13*13
#define LDS_STRIDE 40      // 32 data + 8 pad bf16 per LDS tile row (bank spread)

#if __has_builtin(__builtin_amdgcn_global_load_async_to_lds_b128)
#define HAVE_ASYNC_LDS 1
#else
#define HAVE_ASYNC_LDS 0
#endif

#if HAVE_ASYNC_LDS
#if __has_builtin(__builtin_amdgcn_s_wait_asynccnt)
#define WAIT_ASYNC(n) __builtin_amdgcn_s_wait_asynccnt(n)
#else
#define WAIT_ASYNC(n) asm volatile("s_wait_asynccnt %0" ::"n"(n) : "memory")
#endif
#endif

// ----------------------------- fragment loads ------------------------------
// A fragment: 16x32 bf16 tile, row-major source [rows][ld].
// lane<16 : row=lane,    K = {0..7} in q[0], {16..23} in q[1]
// lane>=16: row=lane-16, K = {8..15} in q[0], {24..31} in q[1]
__device__ inline bf16x16 load_a_frag(const __bf16* base, int ld, int row0,
                                      int k0, int lane) {
    FragU f;
    int r  = lane & 15;
    int hi = (lane >> 4) & 1;
    const __bf16* p = base + (size_t)(row0 + r) * ld + k0 + hi * 8;
    f.q[0] = *(const U4*)p;
    f.q[1] = *(const U4*)(p + 16);
    return f.v;
}

// B fragment: 32x16 bf16 tile; source transposed [cols][ld] (K contiguous).
__device__ inline bf16x16 load_b_frag(const __bf16* base, int ld, int col0,
                                      int k0, int lane) {
    FragU f;
    int c  = lane & 15;
    int hi = (lane >> 4) & 1;
    const __bf16* p = base + (size_t)(col0 + c) * ld + k0 + hi * 8;
    f.q[0] = *(const U4*)p;
    f.q[1] = *(const U4*)(p + 16);
    return f.v;
}

__device__ inline f32x8 wmma_bf16(bf16x16 a, bf16x16 b, f32x8 c) {
    return __builtin_amdgcn_wmma_f32_16x16x32_bf16(
        false, a, false, b, (short)0, c, false, false);
}

__device__ inline float gelu_f(float x) {
    float x3 = x * x * x;
    return 0.5f * x * (1.0f + tanhf(0.7978845608028654f * (x + 0.044715f * x3)));
}

// --------------------- global -> LDS tile staging --------------------------
// Stage a 128-row x 32-col bf16 tile into LDS (row stride LDS_STRIDE).
// 512 x 16B chunks, 256 threads -> 2 chunks each.
__device__ inline void stage_tile(const __bf16* gbase, int ld, int row0, int k0,
                                  __bf16* lds, int tid) {
    #pragma unroll
    for (int it = 0; it < 2; ++it) {
        int c   = tid + it * 256;
        int row = c >> 2;
        int seg = c & 3;
        const __bf16* g = gbase + (size_t)(row0 + row) * ld + k0 + seg * 8;
        __bf16* l = lds + row * LDS_STRIDE + seg * 8;
#if HAVE_ASYNC_LDS
        __builtin_amdgcn_global_load_async_to_lds_b128(
            (v4i*)(void*)g, (v4i*)l, 0, 0);
#else
        *(U4*)l = *(const U4*)g;
#endif
    }
}

// ------------------------------- GEMM kernel -------------------------------
// 256 threads = 8 waves. Block tile 128x128 (double-buffered LDS staging),
// wave tile 32x64 = 2x4 WMMA tiles -> 8 WMMAs per 32-K step.
// MODE 0: QKV scatter (Ncol=1536) -> qb [B,H,NPAD,64], kb same, vb [B,H,64,NPAD]
// MODE 1: outF = resid + acc                              (fp32, ld=Ncol)
// MODE 2: outH = bf16(gelu(acc + bias[c]))                (bf16, ld=Ncol)
// MODE 3: outF = resid + acc + bias[c]                    (fp32, ld=Ncol)
template <int MODE>
__global__ __launch_bounds__(256) void gemm_bf16_kernel(
    const __bf16* __restrict__ A, const __bf16* __restrict__ BT,
    const int K, const int Ncol,
    const float* __restrict__ bias, const float* __restrict__ resid,
    float* __restrict__ outF, __bf16* __restrict__ outH,
    __bf16* __restrict__ qb, __bf16* __restrict__ kb, __bf16* __restrict__ vb)
{
    __shared__ __align__(16) __bf16 Abuf[2][128 * LDS_STRIDE];
    __shared__ __align__(16) __bf16 Bbuf[2][128 * LDS_STRIDE];

    const int tid  = threadIdx.x;
    const int lane = tid & 31;
    const int wave = tid >> 5;
    const int wr = wave & 3;              // 4 row groups of 32
    const int wc = wave >> 2;             // 2 col groups of 64
    const int rowBlk = blockIdx.x * 128;
    const int colBlk = blockIdx.y * 128;

    f32x8 acc[2][4];
    #pragma unroll
    for (int i = 0; i < 2; ++i)
        #pragma unroll
        for (int j = 0; j < 4; ++j)
            acc[i][j] = (f32x8){0.f, 0.f, 0.f, 0.f, 0.f, 0.f, 0.f, 0.f};

    // preload buffer 0
    stage_tile(A,  K, rowBlk, 0, &Abuf[0][0], tid);
    stage_tile(BT, K, colBlk, 0, &Bbuf[0][0], tid);

    const int nk = K >> 5;
    for (int kt = 0; kt < nk; ++kt) {
        const int cur = kt & 1;
        if (kt + 1 < nk) {
            stage_tile(A,  K, rowBlk, (kt + 1) * 32, &Abuf[cur ^ 1][0], tid);
            stage_tile(BT, K, colBlk, (kt + 1) * 32, &Bbuf[cur ^ 1][0], tid);
#if HAVE_ASYNC_LDS
            WAIT_ASYNC(8);      // at most 8 async loads left in flight (each buffer issues 4)
        } else {
            WAIT_ASYNC(0);
#endif
        }
        __syncthreads();

        bf16x16 a0 = load_a_frag(&Abuf[cur][0], LDS_STRIDE, wr * 32,      0, lane);
        bf16x16 a1 = load_a_frag(&Abuf[cur][0], LDS_STRIDE, wr * 32 + 16, 0, lane);
        bf16x16 b0 = load_b_frag(&Bbuf[cur][0], LDS_STRIDE, wc * 64,      0, lane);
        bf16x16 b1 = load_b_frag(&Bbuf[cur][0], LDS_STRIDE, wc * 64 + 16, 0, lane);
        bf16x16 b2 = load_b_frag(&Bbuf[cur][0], LDS_STRIDE, wc * 64 + 32, 0, lane);
        bf16x16 b3 = load_b_frag(&Bbuf[cur][0], LDS_STRIDE, wc * 64 + 48, 0, lane);

        acc[0][0] = wmma_bf16(a0, b0, acc[0][0]);
        acc[0][1] = wmma_bf16(a0, b1, acc[0][1]);
        acc[0][2] = wmma_bf16(a0, b2, acc[0][2]);
        acc[0][3] = wmma_bf16(a0, b3, acc[0][3]);
        acc[1][0] = wmma_bf16(a1, b0, acc[1][0]);
        acc[1][1] = wmma_bf16(a1, b1, acc[1][1]);
        acc[1][2] = wmma_bf16(a1, b2, acc[1][2]);
        acc[1][3] = wmma_bf16(a1, b3, acc[1][3]);

        __syncthreads();
    }

    const int rOff = ((lane >> 4) & 1) * 8;
    const int cL   = lane & 15;
    #pragma unroll
    for (int mi = 0; mi < 2; ++mi) {
        #pragma unroll
        for (int ni = 0; ni < 4; ++ni) {
            #pragma unroll
            for (int v = 0; v < 8; ++v) {
                const int row = rowBlk + wr * 32 + mi * 16 + rOff + v;
                const int col = colBlk + wc * 64 + ni * 16 + cL;
                float val = acc[mi][ni][v];
                if constexpr (MODE == 0) {
                    int bIdx = row / NTOK;
                    int tok  = row - bIdx * NTOK;
                    int sel  = col >> 9;          // 0=q 1=k 2=v
                    int cc   = col & 511;
                    int hh   = cc >> 6;
                    int dh   = cc & 63;
                    size_t bh = (size_t)bIdx * NHEAD + hh;
                    if (sel == 0) {
                        qb[(bh * NPAD + tok) * DHEAD + dh] = (__bf16)val;
                    } else if (sel == 1) {
                        kb[(bh * NPAD + tok) * DHEAD + dh] = (__bf16)val;
                    } else {
                        vb[(bh * DHEAD + dh) * NPAD + tok] = (__bf16)val;
                    }
                } else if constexpr (MODE == 1) {
                    size_t idx = (size_t)row * Ncol + col;
                    outF[idx] = resid[idx] + val;
                } else if constexpr (MODE == 2) {
                    size_t idx = (size_t)row * Ncol + col;
                    outH[idx] = (__bf16)gelu_f(val + bias[col]);
                } else {
                    size_t idx = (size_t)row * Ncol + col;
                    outF[idx] = resid[idx] + val + bias[col];
                }
            }
        }
    }
}

// ----------------------------- attention kernel ----------------------------
// grid (22, H, B); 256 threads = 8 waves. One 16-query tile per block.
// biasPad is pre-padded [H][NPAD][NPAD] (zeros in pad) -> branch-free epilogue.
// Phase 3 uses split-K across all 8 waves with LDS reduction.
__global__ __launch_bounds__(256) void attn_kernel(
    const __bf16* __restrict__ qb, const __bf16* __restrict__ kb,
    const __bf16* __restrict__ vb, const float* __restrict__ biasPad,
    __bf16* __restrict__ obuf)
{
    __shared__ __align__(16) float  S[16][NPAD];
    __shared__ __align__(16) __bf16 P[16][NPAD];

    const int qt   = blockIdx.x;
    const int h    = blockIdx.y;
    const int bIdx = blockIdx.z;
    const int lane = threadIdx.x & 31;
    const int wave = threadIdx.x >> 5;

    const size_t bh = (size_t)bIdx * NHEAD + h;
    const __bf16* qbase = qb + bh * NPAD * DHEAD;
    const __bf16* kbase = kb + bh * NPAD * DHEAD;
    const __bf16* vbase = vb + bh * DHEAD * NPAD;
    const float*  bbase = biasPad + (size_t)h * NPAD * NPAD + (size_t)qt * 16 * NPAD;

    // ---- phase 1: S = q k^T * scale + bias ----
    for (int ct = wave; ct < 22; ct += 8) {
        f32x8 acc = (f32x8){0.f, 0.f, 0.f, 0.f, 0.f, 0.f, 0.f, 0.f};
        #pragma unroll
        for (int k0 = 0; k0 < DHEAD; k0 += 32) {
            bf16x16 a = load_a_frag(qbase, DHEAD, qt * 16, k0, lane);
            bf16x16 b = load_b_frag(kbase, DHEAD, ct * 16, k0, lane);
            acc = wmma_bf16(a, b, acc);
        }
        const int rOff = ((lane >> 4) & 1) * 8;
        const int cL   = lane & 15;
        const int j    = ct * 16 + cL;
        #pragma unroll
        for (int v = 0; v < 8; ++v) {
            S[rOff + v][j] = acc[v] * 0.125f + bbase[(size_t)(rOff + v) * NPAD + j];
        }
    }
    __syncthreads();

    // ---- phase 2: softmax over valid columns, write bf16 P (zero pad) ----
    const int r0 = wave * 2;
    for (int rr = r0; rr < r0 + 2; ++rr) {
        const int i = qt * 16 + rr;
        float m = -3.0e38f;
        for (int j = lane; j < NTOK; j += 32) m = fmaxf(m, S[rr][j]);
        #pragma unroll
        for (int o = 16; o > 0; o >>= 1) m = fmaxf(m, __shfl_xor(m, o));
        float s = 0.f;
        for (int j = lane; j < NTOK; j += 32) {
            float e = __expf(S[rr][j] - m);
            S[rr][j] = e;
            s += e;
        }
        #pragma unroll
        for (int o = 16; o > 0; o >>= 1) s += __shfl_xor(s, o);
        float inv = 1.0f / s;
        for (int j = lane; j < NPAD; j += 32) {
            float p = (j < NTOK && i < NTOK) ? S[rr][j] * inv : 0.f;
            P[rr][j] = (__bf16)p;
        }
    }
    __syncthreads();   // S is dead after this point; reused as reduction buffer

    // ---- phase 3: O = P @ V, split-K over 2 wave groups + LDS reduce ----
    {
        const int dhT  = wave & 3;            // which 16-wide dh tile
        const int half = wave >> 2;           // K-range half
        const int kbeg = half ? 192 : 0;
        const int kend = half ? NPAD : 192;
        f32x8 acc = (f32x8){0.f, 0.f, 0.f, 0.f, 0.f, 0.f, 0.f, 0.f};
        for (int k0 = kbeg; k0 < kend; k0 += 32) {
            bf16x16 a = load_a_frag(&P[0][0], NPAD, 0, k0, lane);
            bf16x16 b = load_b_frag(vbase, NPAD, dhT * 16, k0, lane);
            acc = wmma_bf16(a, b, acc);
        }
        const int rOff = ((lane >> 4) & 1) * 8;
        const int cL   = lane & 15;
        float* Sf = &S[0][0];
        if (half) {
            #pragma unroll
            for (int v = 0; v < 8; ++v)
                Sf[dhT * 256 + (rOff + v) * 16 + cL] = acc[v];
        }
        __syncthreads();
        if (!half) {
            #pragma unroll
            for (int v = 0; v < 8; ++v) {
                const int i = qt * 16 + rOff + v;
                float o = acc[v] + Sf[dhT * 256 + (rOff + v) * 16 + cL];
                if (i < NTOK) {
                    obuf[((size_t)bIdx * NTOK + i) * DMODEL
                         + h * DHEAD + dhT * 16 + cL] = (__bf16)o;
                }
            }
        }
    }
}

// ------------------------------ LayerNorm ----------------------------------
template <bool BF16OUT>
__global__ __launch_bounds__(256) void ln_kernel(
    const float* __restrict__ x, const float* __restrict__ g,
    const float* __restrict__ bta, __bf16* __restrict__ outH,
    float* __restrict__ outF)
{
    const int lane = threadIdx.x & 31;
    const int wave = threadIdx.x >> 5;
    const size_t row = (size_t)blockIdx.x * 8 + wave;
    const float* p = x + row * DMODEL + lane * 16;

    float v[16];
    #pragma unroll
    for (int t = 0; t < 4; ++t) {
        F4 f = *(const F4*)(p + t * 4);
        v[t * 4 + 0] = f.x; v[t * 4 + 1] = f.y;
        v[t * 4 + 2] = f.z; v[t * 4 + 3] = f.w;
    }
    float s = 0.f;
    #pragma unroll
    for (int e = 0; e < 16; ++e) s += v[e];
    #pragma unroll
    for (int o = 16; o > 0; o >>= 1) s += __shfl_xor(s, o);
    const float mu = s * (1.0f / DMODEL);
    float ss = 0.f;
    #pragma unroll
    for (int e = 0; e < 16; ++e) { float c = v[e] - mu; ss += c * c; }
    #pragma unroll
    for (int o = 16; o > 0; o >>= 1) ss += __shfl_xor(ss, o);
    const float rinv = rsqrtf(ss * (1.0f / DMODEL) + 1e-5f);

    const int c0 = lane * 16;
    #pragma unroll
    for (int e = 0; e < 16; ++e) {
        float ov = (v[e] - mu) * rinv * g[c0 + e] + bta[c0 + e];
        if constexpr (BF16OUT) outH[row * DMODEL + c0 + e] = (__bf16)ov;
        else                   outF[row * DMODEL + c0 + e] = ov;
    }
}

// -------------------- relative-position-bias MLP + gather ------------------
__global__ __launch_bounds__(256) void cpb_kernel(
    const float* __restrict__ rel_table, const float* __restrict__ w1,
    const float* __restrict__ b1, const float* __restrict__ w2,
    float* __restrict__ t2)
{
    __shared__ float sred[8][8];
    const int t    = blockIdx.x;
    const int tid  = threadIdx.x;
    const int lane = tid & 31;
    const int wave = tid >> 5;

    float rt[8];
    #pragma unroll
    for (int r = 0; r < 8; ++r) rt[r] = rel_table[t * 8 + r];

    float loc[8] = {0.f, 0.f, 0.f, 0.f, 0.f, 0.f, 0.f, 0.f};
    for (int c = tid; c < 512; c += 256) {
        float a = b1[c];
        #pragma unroll
        for (int r = 0; r < 8; ++r) a += rt[r] * w1[r * 512 + c];
        a = fmaxf(a, 0.f);
        #pragma unroll
        for (int h = 0; h < 8; ++h) loc[h] += a * w2[c * 8 + h];
    }
    #pragma unroll
    for (int h = 0; h < 8; ++h)
        #pragma unroll
        for (int o = 16; o > 0; o >>= 1) loc[h] += __shfl_xor(loc[h], o);
    if (lane == 0)
        #pragma unroll
        for (int h = 0; h < 8; ++h) sred[wave][h] = loc[h];
    __syncthreads();
    if (tid < 8) {
        float s = 0.f;
        #pragma unroll
        for (int w = 0; w < 8; ++w) s += sred[w][tid];
        t2[t * 8 + tid] = s;
    }
}

// gather into PADDED bias [H][NPAD][NPAD]; pad region = 0
__global__ __launch_bounds__(256) void gather_bias_kernel(
    const float* __restrict__ t2, const int* __restrict__ rel_index,
    float* __restrict__ biasPad)
{
    const int totalPad = NPAD * NPAD;
    int ij = blockIdx.x * 256 + threadIdx.x;
    if (ij >= totalPad) return;
    int i = ij / NPAD;
    int j = ij - i * NPAD;
    bool valid = (i < NTOK) && (j < NTOK);
    int id = valid ? rel_index[i * NTOK + j] : 0;
    #pragma unroll
    for (int h = 0; h < 8; ++h) {
        float v = valid ? t2[id * 8 + h] : 0.f;
        biasPad[(size_t)h * totalPad + ij] = v;
    }
}

// ----------------------------- small utilities -----------------------------
__global__ __launch_bounds__(256) void zero_kernel(U4* p, size_t n) {
    size_t i = (size_t)blockIdx.x * 256 + threadIdx.x;
    if (i < n) { U4 z; z.x = 0; z.y = 0; z.z = 0; z.w = 0; p[i] = z; }
}

// Tiled transpose + fp32->bf16 convert: W [K][Ncol] -> WT [Ncol][K].
// 32x32 tiles via LDS, coalesced loads and stores. K, Ncol multiples of 32.
__global__ __launch_bounds__(256) void convT_kernel(
    const float* __restrict__ W, __bf16* __restrict__ WT,
    const int K, const int Ncol)
{
    __shared__ __bf16 tile[32][33];
    const int tx = threadIdx.x & 31;
    const int ty = threadIdx.x >> 5;      // 0..7
    const int k0 = blockIdx.x * 32;
    const int c0 = blockIdx.y * 32;
    #pragma unroll
    for (int t = 0; t < 4; ++t) {
        int r = ty + t * 8;
        tile[r][tx] = (__bf16)W[(size_t)(k0 + r) * Ncol + c0 + tx];
    }
    __syncthreads();
    #pragma unroll
    for (int t = 0; t < 4; ++t) {
        int r = ty + t * 8;
        WT[(size_t)(c0 + r) * K + k0 + tx] = tile[tx][r];
    }
}

// ------------------------------- launcher ----------------------------------
extern "C" void kernel_launch(void* const* d_in, const int* in_sizes, int n_in,
                              void* d_out, int out_size, void* d_ws, size_t ws_size,
                              hipStream_t stream) {
    (void)in_sizes; (void)n_in; (void)out_size; (void)ws_size;

    const float* x_in      = (const float*)d_in[0];
    const float* Wq        = (const float*)d_in[1];
    const float* Wk        = (const float*)d_in[2];
    const float* Wv        = (const float*)d_in[3];
    const float* Wo        = (const float*)d_in[4];
    const float* W1ff      = (const float*)d_in[5];
    const float* b1ff      = (const float*)d_in[6];
    const float* W2ff      = (const float*)d_in[7];
    const float* b2ff      = (const float*)d_in[8];
    const float* ln1_g     = (const float*)d_in[9];
    const float* ln1_b     = (const float*)d_in[10];
    const float* ln2_g     = (const float*)d_in[11];
    const float* ln2_b     = (const float*)d_in[12];
    const float* lnf_g     = (const float*)d_in[13];
    const float* lnf_b     = (const float*)d_in[14];
    const float* rel_table = (const float*)d_in[15];
    const float* cpb_w1    = (const float*)d_in[16];
    const float* cpb_b1    = (const float*)d_in[17];
    const float* cpb_w2    = (const float*)d_in[18];
    const int*   rel_index = (const int*)d_in[19];

    char* ws = (char*)d_ws;
    size_t off = 0;
    auto alloc = [&](size_t bytes) -> char* {
        char* r = ws + off;
        off += (bytes + 255) & ~(size_t)255;
        return r;
    };

    const size_t qkvBytes = (size_t)BWIN * NHEAD * NPAD * DHEAD * 2;

    float*  xbuf   = (float*) alloc((size_t)M_TOK * DMODEL * 4);
    __bf16* hbuf   = (__bf16*)alloc((size_t)M_TOK * DMODEL * 2);
    __bf16* wqkvT  = (__bf16*)alloc((size_t)NLAYER * 1536 * DMODEL * 2);
    __bf16* woT    = (__bf16*)alloc((size_t)NLAYER * DMODEL * DMODEL * 2);
    __bf16* w1T    = (__bf16*)alloc((size_t)NLAYER * DFF * DMODEL * 2);
    __bf16* w2T    = (__bf16*)alloc((size_t)NLAYER * DMODEL * DFF * 2);
    __bf16* qb     = (__bf16*)alloc(qkvBytes);
    __bf16* kb     = (__bf16*)alloc(qkvBytes);
    __bf16* vb     = (__bf16*)alloc(qkvBytes);
    __bf16* obuf   = (__bf16*)alloc((size_t)M_TOK * DMODEL * 2);
    __bf16* tbuf   = (__bf16*)alloc((size_t)M_TOK * DFF * 2);
    float*  t2buf  = (float*) alloc((size_t)TREL * NHEAD * 4);
    float*  biasP  = (float*) alloc((size_t)NHEAD * NPAD * NPAD * 4);

    // zero q/k/v (contiguous) so padding regions stay clean
    {
        size_t n16 = (qkvBytes * 3) / 16;
        zero_kernel<<<(unsigned)((n16 + 255) / 256), 256, 0, stream>>>((U4*)qb, n16);
    }
    // x working copy
    (void)hipMemcpyAsync(xbuf, x_in, (size_t)M_TOK * DMODEL * 4,
                         hipMemcpyDeviceToDevice, stream);

    // weight conversion + transpose (bf16, [Ncol][K]); LDS-tiled, coalesced
    for (int l = 0; l < NLAYER; ++l) {
        const size_t wDD = (size_t)DMODEL * DMODEL;
        convT_kernel<<<dim3(DMODEL / 32, DMODEL / 32), 256, 0, stream>>>(
            Wq + l * wDD, wqkvT + (size_t)l * 1536 * DMODEL, DMODEL, DMODEL);
        convT_kernel<<<dim3(DMODEL / 32, DMODEL / 32), 256, 0, stream>>>(
            Wk + l * wDD, wqkvT + (size_t)l * 1536 * DMODEL + wDD, DMODEL, DMODEL);
        convT_kernel<<<dim3(DMODEL / 32, DMODEL / 32), 256, 0, stream>>>(
            Wv + l * wDD, wqkvT + (size_t)l * 1536 * DMODEL + 2 * wDD, DMODEL, DMODEL);
        convT_kernel<<<dim3(DMODEL / 32, DMODEL / 32), 256, 0, stream>>>(
            Wo + l * wDD, woT + (size_t)l * wDD, DMODEL, DMODEL);
        const size_t wDF = (size_t)DMODEL * DFF;
        convT_kernel<<<dim3(DMODEL / 32, DFF / 32), 256, 0, stream>>>(
            W1ff + l * wDF, w1T + (size_t)l * wDF, DMODEL, DFF);
        convT_kernel<<<dim3(DFF / 32, DMODEL / 32), 256, 0, stream>>>(
            W2ff + l * wDF, w2T + (size_t)l * wDF, DFF, DMODEL);
    }

    // relative position bias (shared across layers), padded layout
    cpb_kernel<<<TREL, 256, 0, stream>>>(rel_table, cpb_w1, cpb_b1, cpb_w2, t2buf);
    gather_bias_kernel<<<(NPAD * NPAD + 255) / 256, 256, 0, stream>>>(
        t2buf, rel_index, biasP);

    const dim3 blk(256);
    for (int l = 0; l < NLAYER; ++l) {
        // LN1 -> hbuf (bf16)
        ln_kernel<true><<<M_TOK / 8, blk, 0, stream>>>(
            xbuf, ln1_g + l * DMODEL, ln1_b + l * DMODEL, hbuf, nullptr);
        // QKV projection -> qb/kb/vb
        gemm_bf16_kernel<0><<<dim3(M_TOK / 128, 1536 / 128), blk, 0, stream>>>(
            hbuf, wqkvT + (size_t)l * 1536 * DMODEL, DMODEL, 1536,
            nullptr, nullptr, nullptr, nullptr, qb, kb, vb);
        // attention -> obuf (bf16 [M, D])
        attn_kernel<<<dim3(22, NHEAD, BWIN), blk, 0, stream>>>(
            qb, kb, vb, biasP, obuf);
        // x = x + o @ Wo
        gemm_bf16_kernel<1><<<dim3(M_TOK / 128, DMODEL / 128), blk, 0, stream>>>(
            obuf, woT + (size_t)l * DMODEL * DMODEL, DMODEL, DMODEL,
            nullptr, xbuf, xbuf, nullptr, nullptr, nullptr, nullptr);
        // LN2 -> hbuf
        ln_kernel<true><<<M_TOK / 8, blk, 0, stream>>>(
            xbuf, ln2_g + l * DMODEL, ln2_b + l * DMODEL, hbuf, nullptr);
        // t = gelu(h @ W1 + b1) (bf16)
        gemm_bf16_kernel<2><<<dim3(M_TOK / 128, DFF / 128), blk, 0, stream>>>(
            hbuf, w1T + (size_t)l * DFF * DMODEL, DMODEL, DFF,
            b1ff + l * DFF, nullptr, nullptr, tbuf, nullptr, nullptr, nullptr);
        // x = x + t @ W2 + b2
        gemm_bf16_kernel<3><<<dim3(M_TOK / 128, DMODEL / 128), blk, 0, stream>>>(
            tbuf, w2T + (size_t)l * DMODEL * DFF, DFF, DMODEL,
            b2ff + l * DMODEL, xbuf, xbuf, nullptr, nullptr, nullptr, nullptr);
    }

    // final LayerNorm -> d_out (fp32)
    ln_kernel<false><<<M_TOK / 8, blk, 0, stream>>>(
        xbuf, lnf_g, lnf_b, nullptr, (float*)d_out);
}
